// ActionDetectionModel_55061480735413
// MI455X (gfx1250) — matrix operand order repaired and checked by reading, not compiled, not run on hardware
//
#include <hip/hip_runtime.h>
#include <stdint.h>

// ---------------------------------------------------------------------------
// CDNA5 (gfx1250) types for WMMA bf16: D(f32 16x16) = A(16x32 bf16) x B(32x16 bf16) + C
// ---------------------------------------------------------------------------
typedef __bf16 bf16_t;
typedef bf16_t v16bf __attribute__((ext_vector_type(16)));
typedef float  v8f   __attribute__((ext_vector_type(8)));
typedef unsigned int v4u __attribute__((ext_vector_type(4)));

union FragB16 { v16bf v; v4u q[2]; };

__device__ __forceinline__ unsigned short f2bf(float f) {
  unsigned int u = __float_as_uint(f);
  u += 0x7FFFu + ((u >> 16) & 1u);   // round-to-nearest-even
  return (unsigned short)(u >> 16);
}
__device__ __forceinline__ float bf2f(unsigned short h) {
  return __uint_as_float(((unsigned int)h) << 16);
}

__device__ __forceinline__ v8f wmma_bf16(const FragB16& a, const FragB16& b, v8f c) {
  return __builtin_amdgcn_wmma_f32_16x16x32_bf16(false, a.v, false, b.v, (short)0, c,
                                                 false, false);
}
// A-fragment (16x32): lane holds K chunks {alo..alo+7, alo+16..alo+23}, alo = lane<16?0:8
__device__ __forceinline__ FragB16 loadA(const unsigned short* row, int kc, int alo) {
  FragB16 t;
  t.q[0] = *(const v4u*)(row + kc + alo);
  t.q[1] = *(const v4u*)(row + kc + alo + 16);
  return t;
}
// B-fragment (32x16, stored transposed [N][K]): lane holds K chunks {blo..blo+7, blo+16..blo+23}, blo = lane<16?0:8
__device__ __forceinline__ FragB16 loadB(const unsigned short* row, int kc, int blo) {
  FragB16 t;
  t.q[0] = *(const v4u*)(row + kc + blo);
  t.q[1] = *(const v4u*)(row + kc + blo + 16);
  return t;
}

// ---------------------------------------------------------------------------
// Elementwise converts / weight transposes
// ---------------------------------------------------------------------------
__global__ void k_cvt_f32_bf16(const float* __restrict__ a, unsigned short* __restrict__ b, int n) {
  int i = blockIdx.x * blockDim.x + threadIdx.x;
  if (i < n) b[i] = f2bf(a[i]);
}

// w: [3][3][cin][cout] f32  ->  wt: [9][cout][cin] bf16
__global__ void k_wt_vgg(const float* __restrict__ w, unsigned short* __restrict__ wt,
                         int cin, int cout) {
  int idx = blockIdx.x * blockDim.x + threadIdx.x;
  int total = 9 * cin * cout;
  if (idx >= total) return;
  int co = idx % cout;
  int rest = idx / cout;
  int ci = rest % cin;
  int k9 = rest / cin;
  wt[((size_t)k9 * cout + co) * cin + ci] = f2bf(w[((size_t)k9 * cin + ci) * cout + co]);
}

// a: [K][N] f32 -> at: [N][K] bf16
__global__ void k_wt_mat(const float* __restrict__ a, unsigned short* __restrict__ at,
                         int K, int N) {
  int idx = blockIdx.x * blockDim.x + threadIdx.x;
  if (idx >= K * N) return;
  int n = idx % N;
  int k = idx / N;
  at[(size_t)n * K + k] = f2bf(a[(size_t)k * N + n]);
}

// Zero the 1-px halo ring of a padded activation buffer [F][H+2][W+2][C]
__global__ void k_zero_halo(unsigned short* __restrict__ x, int F, int H, int W, int C) {
  int ring = 2 * (W + 2) + 2 * H;
  int idx = blockIdx.x * blockDim.x + threadIdx.x;
  int total = F * ring * C;
  if (idx >= total) return;
  int c = idx % C;
  int rest = idx / C;
  int ri = rest % ring;
  int f = rest / ring;
  int y, xx;
  if (ri < W + 2) { y = 0; xx = ri; }
  else if (ri < 2 * (W + 2)) { y = H + 1; xx = ri - (W + 2); }
  else {
    int i = ri - 2 * (W + 2);
    y = (i % H) + 1;
    xx = (i / H) ? (W + 1) : 0;
  }
  x[(((size_t)f * (H + 2) + y) * (W + 2) + xx) * C + c] = 0;
}

// Extract final features: padded [128][4][4][512] interior (2x2) -> flat [128][2048]
__global__ void k_extract_feats(const unsigned short* __restrict__ x, unsigned short* __restrict__ y) {
  int idx = blockIdx.x * blockDim.x + threadIdx.x;
  if (idx >= 128 * 2 * 2 * 512) return;
  int c = idx & 511;
  int rest = idx >> 9;
  int w = rest & 1; rest >>= 1;
  int h = rest & 1;
  int f = rest >> 1;
  y[idx] = x[(((size_t)f * 4 + h + 1) * 4 + w + 1) * 512 + c];
}

// ---------------------------------------------------------------------------
// Layer 0: 3x3 conv, Cin=3 (K=27 not WMMA-friendly; 0.5% of total FLOPs) VALU
// x: [F][64][64][3] bf16 (unpadded), w: [3][3][3][64] f32,
// y: padded [F][66][66][64] bf16 (interior write)
// ---------------------------------------------------------------------------
__global__ void k_conv0(const unsigned short* __restrict__ x, const float* __restrict__ w,
                        const float* __restrict__ bias, unsigned short* __restrict__ y, int F) {
  int idx = blockIdx.x * blockDim.x + threadIdx.x;
  int total = F * 64 * 64 * 64;
  if (idx >= total) return;
  int co = idx & 63;
  int p  = (idx >> 6) & 4095;
  int f  = idx >> 18;
  int h = p >> 6, ww = p & 63;
  float acc = bias[co];
  for (int kh = 0; kh < 3; ++kh) {
    int ih = h + kh - 1;
    if (ih < 0 || ih > 63) continue;
    for (int kw = 0; kw < 3; ++kw) {
      int iw = ww + kw - 1;
      if (iw < 0 || iw > 63) continue;
      const unsigned short* xp = x + (((size_t)f * 64 + ih) * 64 + iw) * 3;
      const float* wp = w + ((kh * 3 + kw) * 3) * 64 + co;
      acc += bf2f(xp[0]) * wp[0] + bf2f(xp[1]) * wp[64] + bf2f(xp[2]) * wp[128];
    }
  }
  acc = acc > 0.f ? acc : 0.f;
  y[(((size_t)f * 66 + h + 1) * 66 + ww + 1) * 64 + co] = f2bf(acc);
}

// ---------------------------------------------------------------------------
// Implicit-GEMM 3x3 SAME conv via WMMA bf16 over zero-padded NHWC input.
// One wave => 2 pixel-tiles (32 px) x 64 output channels: 8 accumulators,
// each B fragment reused by 2 A tiles, each A fragment by 4 B tiles
// (~21 FLOP/byte from cache).  Branch-free inner loop.
// Pixel tiles are paired on the global tile index f*pTiles+pt, so 4x4 layers
// (16 px/frame) pair across frames; F*pTiles is always even.
// x:  padded [F][H+2][W+2][Cin] bf16
// wt: [9][Cout][Cin] bf16
// y:  padded [F][H+2][W+2][Cout] bf16 (interior write)
// Cin mult 32, Cout mult 64, H*W mult 16.
// ---------------------------------------------------------------------------
__global__ __launch_bounds__(256)
void k_conv3x3_wmma(const unsigned short* __restrict__ x, const unsigned short* __restrict__ wt,
                    const float* __restrict__ bias, unsigned short* __restrict__ y,
                    int F, int H, int W, int Cin, int Cout) {
  int gw = (blockIdx.x * blockDim.x + threadIdx.x) >> 5;
  int lane = threadIdx.x & 31;
  int pTiles = (H * W) >> 4;
  int pairs = (F * pTiles) >> 1;
  int cG = Cout >> 6;                    // groups of 4 co-tiles
  int total = pairs * cG;
  if (gw >= total) return;               // wave-uniform: EXEC all-1s at WMMAs
  int cg   = gw % cG;
  int pair = gw / cG;
  int gt0 = pair * 2, gt1 = gt0 + 1;
  int f0 = gt0 / pTiles, pt0 = gt0 % pTiles;
  int f1 = gt1 / pTiles, pt1 = gt1 % pTiles;
  int r  = lane & 15;
  int p0 = pt0 * 16 + r;
  int p1 = pt1 * 16 + r;
  int ph0 = p0 / W, pw0 = p0 % W;
  int ph1 = p1 / W, pw1 = p1 % W;
  int alo = (lane < 16) ? 0 : 8;
  int blo = (lane < 16) ? 0 : 8;
  int Hp = H + 2, Wp = W + 2;
  v8f acc[8] = {};
  for (int kh = 0; kh < 3; ++kh) {
    for (int kw = 0; kw < 3; ++kw) {
      const unsigned short* xrow0 = x + (((size_t)f0 * Hp + ph0 + kh) * Wp + pw0 + kw) * Cin;
      const unsigned short* xrow1 = x + (((size_t)f1 * Hp + ph1 + kh) * Wp + pw1 + kw) * Cin;
      const unsigned short* wrow  = wt + ((size_t)(kh * 3 + kw) * Cout + cg * 64 + r) * Cin;
      for (int kc = 0; kc < Cin; kc += 32) {
        FragB16 a0 = loadA(xrow0, kc, alo);
        FragB16 a1 = loadA(xrow1, kc, alo);
        __builtin_prefetch(wrow + kc + 3 * (size_t)16 * Cin, 0, 3);
#pragma unroll
        for (int rep = 0; rep < 4; ++rep) {
          FragB16 b = loadB(wrow + (size_t)rep * 16 * Cin, kc, blo);
          acc[rep]     = wmma_bf16(a0, b, acc[rep]);
          acc[4 + rep] = wmma_bf16(a1, b, acc[4 + rep]);
        }
      }
    }
  }
  int mb = (lane >> 4) << 3;
  float bv[4];
#pragma unroll
  for (int rep = 0; rep < 4; ++rep) bv[rep] = bias[cg * 64 + rep * 16 + r];
#pragma unroll
  for (int t = 0; t < 2; ++t) {
    int f  = t ? f1 : f0;
    int pt = t ? pt1 : pt0;
#pragma unroll
    for (int i = 0; i < 8; ++i) {
      int pp = pt * 16 + mb + i;
      int hh = pp / W, wo = pp % W;
      size_t obase = (((size_t)f * Hp + hh + 1) * Wp + wo + 1) * Cout + cg * 64 + r;
#pragma unroll
      for (int rep = 0; rep < 4; ++rep) {
        float v = acc[t * 4 + rep][i] + bv[rep];
        v = v > 0.f ? v : 0.f;
        y[obase + rep * 16] = f2bf(v);
      }
    }
  }
}

// ---------------------------------------------------------------------------
// 2x2 maxpool stride 2, padded NHWC bf16 -> padded NHWC bf16 (interior)
// ---------------------------------------------------------------------------
__global__ void k_maxpool(const unsigned short* __restrict__ x, unsigned short* __restrict__ y,
                          int F, int Ho, int Wo, int C) {
  int idx = blockIdx.x * blockDim.x + threadIdx.x;
  int total = F * Ho * Wo * C;
  if (idx >= total) return;
  int c = idx % C;
  int rest = idx / C;
  int wo = rest % Wo; rest /= Wo;
  int ho = rest % Ho;
  int f  = rest / Ho;
  int Hp = Ho * 2 + 2, Wp = Wo * 2 + 2;   // input padded dims
  size_t base = (((size_t)f * Hp + 2 * ho + 1) * Wp + 2 * wo + 1) * C + c;
  float a = bf2f(x[base]);
  float b = bf2f(x[base + C]);
  float d = bf2f(x[base + (size_t)Wp * C]);
  float e = bf2f(x[base + (size_t)Wp * C + C]);
  float m = fmaxf(fmaxf(a, b), fmaxf(d, e));
  y[(((size_t)f * (Ho + 2) + ho + 1) * (Wo + 2) + wo + 1) * C + c] = f2bf(m);
}

// ---------------------------------------------------------------------------
// WMMA GEMM: C[M][N] f32 = A[M][K] bf16 * Bt[N][K] bf16
// Wave => 32 x 64 tile (2 m-tiles x 4 n-tiles, 8 accumulators).
// K mult 32, N mult 64, M mult 32.
// ---------------------------------------------------------------------------
__global__ __launch_bounds__(256)
void k_gemm(const unsigned short* __restrict__ A, const unsigned short* __restrict__ Bt,
            float* __restrict__ C, int M, int N, int K) {
  int gw = (blockIdx.x * blockDim.x + threadIdx.x) >> 5;
  int lane = threadIdx.x & 31;
  int nG = N >> 6;
  int mG = M >> 5;
  int total = mG * nG;
  if (gw >= total) return;
  int mg = gw / nG, ng = gw % nG;
  int r = lane & 15;
  int alo = (lane < 16) ? 0 : 8;
  int blo = (lane < 16) ? 0 : 8;
  const unsigned short* arow0 = A + (size_t)(mg * 32 + r) * K;
  const unsigned short* arow1 = arow0 + (size_t)16 * K;
  const unsigned short* brow  = Bt + (size_t)(ng * 64 + r) * K;
  v8f acc[8] = {};
  for (int kc = 0; kc < K; kc += 32) {
    FragB16 a0 = loadA(arow0, kc, alo);
    FragB16 a1 = loadA(arow1, kc, alo);
    __builtin_prefetch(brow + kc + 512, 0, 3);
#pragma unroll
    for (int rep = 0; rep < 4; ++rep) {
      FragB16 b = loadB(brow + (size_t)rep * 16 * K, kc, blo);
      acc[rep]     = wmma_bf16(a0, b, acc[rep]);
      acc[4 + rep] = wmma_bf16(a1, b, acc[4 + rep]);
    }
  }
  int mb = (lane >> 4) << 3;
#pragma unroll
  for (int t = 0; t < 2; ++t) {
#pragma unroll
    for (int i = 0; i < 8; ++i) {
      size_t cbase = (size_t)(mg * 32 + t * 16 + mb + i) * N + ng * 64 + r;
#pragma unroll
      for (int rep = 0; rep < 4; ++rep) C[cbase + rep * 16] = acc[t * 4 + rep][i];
    }
  }
}

// ---------------------------------------------------------------------------
// Peephole BiLSTM scan (one direction per block; grid = 2 blocks, 512 thr).
// State h,c resident in LDS; per-step projections via WMMA.
// Each wave owns 4 column tiles of one gate (g = wave>>2), so the LDS A
// fragment (h or c) is loaded once per k-chunk and reused by 4 WMMAs.
// zx: [B=8][T=16][1024] f32 (row = b*16+t); rkt: [1024][256] bf16;
// wpt: [3][256][256] bf16 (Wi^T,Wf^T,Wo^T); seq: [B][T][512] f32.
// ---------------------------------------------------------------------------
#define LSTM_U 256
__global__ __launch_bounds__(512)
void k_lstm_scan(const float* __restrict__ zxf, const float* __restrict__ zxb,
                 const unsigned short* __restrict__ rktf, const unsigned short* __restrict__ rktb,
                 const unsigned short* __restrict__ wptf, const unsigned short* __restrict__ wptb,
                 float* __restrict__ seq) {
  __shared__ __align__(16) unsigned short hS[16 * LSTM_U];   // h bf16 (rows 8..15 = 0)
  __shared__ __align__(16) unsigned short cbS[16 * LSTM_U];  // c bf16 copy for WMMA A
  __shared__ float cS[8 * LSTM_U];                           // c f32 master
  __shared__ float gS[4 * 8 * LSTM_U];                       // gate pre-activations
  const int dir = blockIdx.x;
  const float* zx = dir ? zxb : zxf;
  const unsigned short* rkt = dir ? rktb : rktf;
  const unsigned short* wpt = dir ? wptb : wptf;
  int tid = threadIdx.x, lane = tid & 31, wave = tid >> 5;
  for (int i = tid; i < 16 * LSTM_U; i += 512) { hS[i] = 0; cbS[i] = 0; }
  for (int i = tid; i < 8 * LSTM_U; i += 512) cS[i] = 0.f;
  __syncthreads();
  int alo = (lane < 16) ? 0 : 8;
  int blo = (lane < 16) ? 0 : 8;
  int r = lane & 15;
  int mb = (lane >> 4) << 3;
  int g  = wave >> 2;                 // this wave's gate (0=cell,1=i,2=f,3=o)
  int j0 = wave * 4;                  // first of 4 column tiles in [8 x 1024]
  const unsigned short* zr[4];
  const unsigned short* pr[4];
#pragma unroll
  for (int rep = 0; rep < 4; ++rep) {
    zr[rep] = rkt + (size_t)((j0 + rep) * 16 + r) * LSTM_U;
    int ucol = ((j0 + rep) & 15) << 4;
    pr[rep] = wpt + (size_t)((g >= 1 ? (g - 1) : 0) * LSTM_U + ucol + r) * LSTM_U;
  }
  for (int s = 0; s < 16; ++s) {
    int tIdx = dir ? (15 - s) : s;
    // ---- Phase A: gate pre-activations ----
    v8f acc[4];
#pragma unroll
    for (int rep = 0; rep < 4; ++rep) {
#pragma unroll
      for (int i = 0; i < 8; ++i) {
        int mm = mb + i;
        acc[rep][i] = (mm < 8) ? zx[(size_t)(mm * 16 + tIdx) * 1024 + (j0 + rep) * 16 + r] : 0.f;
      }
    }
    const unsigned short* hrow = hS + r * LSTM_U;
    for (int kc = 0; kc < LSTM_U; kc += 32) {          // h @ rkernel
      FragB16 a = loadA(hrow, kc, alo);
#pragma unroll
      for (int rep = 0; rep < 4; ++rep) {
        FragB16 b = loadB(zr[rep], kc, blo);
        acc[rep] = wmma_bf16(a, b, acc[rep]);
      }
    }
    if (g >= 1) {                                       // peephole c @ W{i,f,o}
      const unsigned short* crow = cbS + r * LSTM_U;
      for (int kc = 0; kc < LSTM_U; kc += 32) {
        FragB16 a = loadA(crow, kc, alo);
#pragma unroll
        for (int rep = 0; rep < 4; ++rep) {
          FragB16 b = loadB(pr[rep], kc, blo);
          acc[rep] = wmma_bf16(a, b, acc[rep]);
        }
      }
    }
#pragma unroll
    for (int rep = 0; rep < 4; ++rep) {
      int ucol = ((j0 + rep) & 15) << 4;
#pragma unroll
      for (int i = 0; i < 8; ++i) {
        int mm = mb + i;
        if (mm < 8) gS[(size_t)(g * 8 + mm) * LSTM_U + ucol + r] = acc[rep][i];
      }
    }
    __syncthreads();
    // ---- Phase B: elementwise gate math + state update ----
#pragma unroll
    for (int rr = 0; rr < 4; ++rr) {
      int idx = tid + rr * 512;        // 0..2047 = (b,u)
      int b = idx >> 8, u = idx & 255;
      float z0 = gS[(0 * 8 + b) * LSTM_U + u];
      float zi = gS[(1 * 8 + b) * LSTM_U + u];
      float zf = gS[(2 * 8 + b) * LSTM_U + u];
      float zo = gS[(3 * 8 + b) * LSTM_U + u];
      float c  = cS[b * LSTM_U + u];
      float ig = 1.f / (1.f + __expf(-zi));
      float fg = 1.f / (1.f + __expf(-zf));
      float og = 1.f / (1.f + __expf(-zo));
      float cn = fg * c + ig * tanhf(z0);
      float hh = og * tanhf(cn);
      cS[b * LSTM_U + u] = cn;
      cbS[b * LSTM_U + u] = f2bf(cn);
      hS[b * LSTM_U + u] = f2bf(hh);
      seq[(size_t)(b * 16 + tIdx) * 512 + dir * LSTM_U + u] = hh;
    }
    __syncthreads();
  }
}

// ---------------------------------------------------------------------------
// Dense head
// ---------------------------------------------------------------------------
__global__ void k_dense1(const float* __restrict__ seq2, const float* __restrict__ w1,
                         const float* __restrict__ b1, float* __restrict__ out) {
  int j = blockIdx.x * blockDim.x + threadIdx.x;
  if (j >= 8 * 1024) return;
  int b = j >> 10, col = j & 1023;
  float acc = b1[col];
  const float* hf = seq2 + (size_t)(b * 16 + 15) * 512;        // forward last
  const float* hb = seq2 + (size_t)(b * 16 + 0) * 512 + 256;   // backward last
  for (int k = 0; k < 256; ++k) acc += hf[k] * w1[(size_t)k * 1024 + col];
  for (int k = 0; k < 256; ++k) acc += hb[k] * w1[(size_t)(256 + k) * 1024 + col];
  out[b * 1024 + col] = acc > 0.f ? acc : 0.f;
}

__global__ __launch_bounds__(128)
void k_dense2_softmax(const float* __restrict__ h, const float* __restrict__ w2,
                      const float* __restrict__ b2, float* __restrict__ out) {
  int b = blockIdx.x, tid = threadIdx.x;
  __shared__ float red[128];
  __shared__ float logits[101];
  if (tid < 101) {
    float acc = b2[tid];
    const float* hb = h + (size_t)b * 1024;
    for (int k = 0; k < 1024; ++k) acc += hb[k] * w2[(size_t)k * 101 + tid];
    logits[tid] = acc;
  }
  __syncthreads();
  red[tid] = (tid < 101) ? logits[tid] : -1e30f;
  __syncthreads();
  for (int s = 64; s > 0; s >>= 1) { if (tid < s) red[tid] = fmaxf(red[tid], red[tid + s]); __syncthreads(); }
  float mx = red[0];
  __syncthreads();
  red[tid] = (tid < 101) ? __expf(logits[tid] - mx) : 0.f;
  __syncthreads();
  for (int s = 64; s > 0; s >>= 1) { if (tid < s) red[tid] += red[tid + s]; __syncthreads(); }
  float sum = red[0];
  if (tid < 101) out[b * 101 + tid] = __expf(logits[tid] - mx) / sum;
}

// ---------------------------------------------------------------------------
// Orchestration
// ---------------------------------------------------------------------------
extern "C" void kernel_launch(void* const* d_in, const int* in_sizes, int n_in,
                              void* d_out, int out_size, void* d_ws, size_t ws_size,
                              hipStream_t stream) {
  (void)in_sizes; (void)n_in; (void)out_size; (void)ws_size;
  const int F = 128;                                    // B*T frames
  static const int  cins[16]  = {3,64,64,128,128,256,256,256,256,512,512,512,512,512,512,512};
  static const int  couts[16] = {64,64,128,128,256,256,256,256,512,512,512,512,512,512,512,512};
  static const int  hw[16]    = {64,64,32,32,16,16,16,16,8,8,8,8,4,4,4,4};
  static const bool pool[16]  = {false,true,false,true,false,false,false,true,
                                 false,false,false,true,false,false,false,true};

  char* ws = (char*)d_ws;
  size_t off = 0;
  auto alloc = [&](size_t bytes) -> void* {
    off = (off + 255) & ~(size_t)255;
    void* p = ws + off;
    off += bytes;
    return p;
  };

  unsigned short* frames = (unsigned short*)alloc((size_t)F * 64 * 64 * 3 * 2);
  unsigned short* actA   = (unsigned short*)alloc((size_t)F * 66 * 66 * 64 * 2);  // padded
  unsigned short* actB   = (unsigned short*)alloc((size_t)F * 66 * 66 * 64 * 2);  // padded
  unsigned short* feats  = (unsigned short*)alloc((size_t)F * 2048 * 2);
  size_t wtElems = 0, wtOff[16] = {};
  for (int i = 1; i < 16; ++i) { wtOff[i] = wtElems; wtElems += (size_t)9 * cins[i] * couts[i]; }
  unsigned short* wtbuf = (unsigned short*)alloc(wtElems * 2);

  unsigned short* kt1f  = (unsigned short*)alloc((size_t)1024 * 2048 * 2);
  unsigned short* kt1b  = (unsigned short*)alloc((size_t)1024 * 2048 * 2);
  unsigned short* kt2f  = (unsigned short*)alloc((size_t)1024 * 512 * 2);
  unsigned short* kt2b  = (unsigned short*)alloc((size_t)1024 * 512 * 2);
  unsigned short* rkt1f = (unsigned short*)alloc((size_t)1024 * 256 * 2);
  unsigned short* rkt1b = (unsigned short*)alloc((size_t)1024 * 256 * 2);
  unsigned short* rkt2f = (unsigned short*)alloc((size_t)1024 * 256 * 2);
  unsigned short* rkt2b = (unsigned short*)alloc((size_t)1024 * 256 * 2);
  unsigned short* wpt1f = (unsigned short*)alloc((size_t)3 * 256 * 256 * 2);
  unsigned short* wpt1b = (unsigned short*)alloc((size_t)3 * 256 * 256 * 2);
  unsigned short* wpt2f = (unsigned short*)alloc((size_t)3 * 256 * 256 * 2);
  unsigned short* wpt2b = (unsigned short*)alloc((size_t)3 * 256 * 256 * 2);
  float* zx1f = (float*)alloc((size_t)128 * 1024 * 4);
  float* zx1b = (float*)alloc((size_t)128 * 1024 * 4);
  float* zx2f = (float*)alloc((size_t)128 * 1024 * 4);
  float* zx2b = (float*)alloc((size_t)128 * 1024 * 4);
  float* seq1 = (float*)alloc((size_t)128 * 512 * 4);
  float* seq2 = (float*)alloc((size_t)128 * 512 * 4);
  unsigned short* seq1b = (unsigned short*)alloc((size_t)128 * 512 * 2);
  float* den1 = (float*)alloc((size_t)8 * 1024 * 4);

  // --- weight/data preprocessing (bf16 + transposes) ---
  { int n = F * 64 * 64 * 3;
    k_cvt_f32_bf16<<<(n + 255) / 256, 256, 0, stream>>>((const float*)d_in[0], frames, n); }
  for (int i = 1; i < 16; ++i) {
    int n = 9 * cins[i] * couts[i];
    k_wt_vgg<<<(n + 255) / 256, 256, 0, stream>>>((const float*)d_in[1 + 2 * i],
                                                  wtbuf + wtOff[i], cins[i], couts[i]);
  }
  auto tmat = [&](int idx, unsigned short* dst, int K, int N) {
    int n = K * N;
    k_wt_mat<<<(n + 255) / 256, 256, 0, stream>>>((const float*)d_in[idx], dst, K, N);
  };
  // lstm1 fwd: d_in[33..37] kernel,rkernel,Wi,Wf,Wo ; lstm1 bwd: 38..42 ; lstm2: 43..47, 48..52
  tmat(33, kt1f, 2048, 1024);  tmat(34, rkt1f, 256, 1024);
  tmat(35, wpt1f + 0 * 65536, 256, 256); tmat(36, wpt1f + 1 * 65536, 256, 256); tmat(37, wpt1f + 2 * 65536, 256, 256);
  tmat(38, kt1b, 2048, 1024);  tmat(39, rkt1b, 256, 1024);
  tmat(40, wpt1b + 0 * 65536, 256, 256); tmat(41, wpt1b + 1 * 65536, 256, 256); tmat(42, wpt1b + 2 * 65536, 256, 256);
  tmat(43, kt2f, 512, 1024);   tmat(44, rkt2f, 256, 1024);
  tmat(45, wpt2f + 0 * 65536, 256, 256); tmat(46, wpt2f + 1 * 65536, 256, 256); tmat(47, wpt2f + 2 * 65536, 256, 256);
  tmat(48, kt2b, 512, 1024);   tmat(49, rkt2b, 256, 1024);
  tmat(50, wpt2b + 0 * 65536, 256, 256); tmat(51, wpt2b + 1 * 65536, 256, 256); tmat(52, wpt2b + 2 * 65536, 256, 256);

  // --- VGG19 conv stack (padded NHWC activations) ---
  { int n = F * 64 * 64 * 64;
    k_conv0<<<(n + 255) / 256, 256, 0, stream>>>(frames, (const float*)d_in[1],
                                                 (const float*)d_in[2], actA, F); }
  unsigned short* cur = actA;
  unsigned short* nxt = actB;
  for (int i = 1; i < 16; ++i) {
    int H = hw[i];
    { // zero the halo ring of the conv input (producer wrote interior only)
      int ring = 2 * (H + 2) + 2 * H;
      int n = F * ring * cins[i];
      k_zero_halo<<<(n + 255) / 256, 256, 0, stream>>>(cur, F, H, H, cins[i]);
    }
    long tiles = ((long)F * ((H * H) >> 4) / 2) * (couts[i] >> 6);
    long thr = tiles * 32;
    k_conv3x3_wmma<<<(int)((thr + 255) / 256), 256, 0, stream>>>(
        cur, wtbuf + wtOff[i], (const float*)d_in[2 + 2 * i], nxt,
        F, H, H, cins[i], couts[i]);
    { unsigned short* t = cur; cur = nxt; nxt = t; }
    if (pool[i]) {
      int Ho = H / 2;
      int n = F * Ho * Ho * couts[i];
      k_maxpool<<<(n + 255) / 256, 256, 0, stream>>>(cur, nxt, F, Ho, Ho, couts[i]);
      unsigned short* t = cur; cur = nxt; nxt = t;
    }
  }
  { int n = 128 * 2048;
    k_extract_feats<<<(n + 255) / 256, 256, 0, stream>>>(cur, feats); }

  auto gemm = [&](const unsigned short* A, const unsigned short* Bt, float* C,
                  int M, int N, int K) {
    int tiles = (M >> 5) * (N >> 6);
    k_gemm<<<(tiles * 32 + 255) / 256, 256, 0, stream>>>(A, Bt, C, M, N, K);
  };

  // --- BiLSTM layer 1 ---
  gemm(feats, kt1f, zx1f, 128, 1024, 2048);
  gemm(feats, kt1b, zx1b, 128, 1024, 2048);
  k_lstm_scan<<<2, 512, 0, stream>>>(zx1f, zx1b, rkt1f, rkt1b, wpt1f, wpt1b, seq1);
  { int n = 128 * 512;
    k_cvt_f32_bf16<<<(n + 255) / 256, 256, 0, stream>>>(seq1, seq1b, n); }

  // --- BiLSTM layer 2 ---
  gemm(seq1b, kt2f, zx2f, 128, 1024, 512);
  gemm(seq1b, kt2b, zx2b, 128, 1024, 512);
  k_lstm_scan<<<2, 512, 0, stream>>>(zx2f, zx2b, rkt2f, rkt2b, wpt2f, wpt2b, seq2);

  // --- Dense head ---
  k_dense1<<<(8 * 1024 + 255) / 256, 256, 0, stream>>>(seq2, (const float*)d_in[53],
                                                       (const float*)d_in[54], den1);
  k_dense2_softmax<<<8, 128, 0, stream>>>(den1, (const float*)d_in[55],
                                          (const float*)d_in[56], (float*)d_out);
}
